// GAT_6081673691660
// MI455X (gfx1250) — matrix operand order repaired and checked by reading, not compiled, not run on hardware
//
#include <hip/hip_runtime.h>
#include <cstdint>
#include <cstddef>

// ---------------------------------------------------------------------------
// Types for CDNA5 WMMA (gfx1250, wave32)
// ---------------------------------------------------------------------------
typedef __bf16 bf16;
typedef __attribute__((ext_vector_type(16))) __bf16   v16bf;
typedef __attribute__((ext_vector_type(8)))  float    v8f;
typedef __attribute__((ext_vector_type(4)))  unsigned u32x4;

#define NEG_SLOPE 0.2f
#define GAT_EPS   1e-16f
// order-preserving float->uint encoding; enc(-inf) = ~0xFF800000 = 0x007FFFFF
#define NEG_INF_KEY 0x007FFFFFu

__device__ __forceinline__ unsigned fenc(float f) {
  unsigned b = __float_as_uint(f);
  return (b & 0x80000000u) ? ~b : (b | 0x80000000u);
}
__device__ __forceinline__ float fdec(unsigned k) {
  unsigned b = (k & 0x80000000u) ? (k & 0x7FFFFFFFu) : ~k;
  return __uint_as_float(b);
}

union Frag { u32x4 q[2]; v16bf v; };

// A fragment (16x32 bf16, MxK): lane = {lm, hi}; holds row M=lm,
// K = kbase + hi*8 + {0..7} and kbase + 16 + hi*8 + {0..7}  (two 16B runs)
__device__ __forceinline__ v16bf load_a_frag(const bf16* rowp, int kbase, int hi) {
  Frag f;
  const bf16* p = rowp + kbase + hi * 8;
  f.q[0] = *(const u32x4*)(p);
  f.q[1] = *(const u32x4*)(p + 16);
  return f.v;
}
// B fragment (32x16 bf16, KxN), column-major weights: lane holds col N=lm,
// K = kbase + hi*8 + {0..7} and kbase + 16 + hi*8 + {0..7}  (two 16B runs)
__device__ __forceinline__ v16bf load_b_frag(const bf16* colp, int kbase, int hi) {
  Frag f;
  const bf16* p = colp + kbase + hi * 8;
  f.q[0] = *(const u32x4*)(p);
  f.q[1] = *(const u32x4*)(p + 16);
  return f.v;
}

// ---------------------------------------------------------------------------
// Prep kernels
// ---------------------------------------------------------------------------
__global__ void cvt_f32_bf16(const float* __restrict__ x, bf16* __restrict__ y,
                             long long n) {
  long long i = blockIdx.x * (long long)blockDim.x + threadIdx.x;
  if (i < n) y[i] = (bf16)x[i];
}

// W1 [512,256] f32 row-major -> w1bt[c*512 + k] bf16 (column-major)
__global__ void prep_w1(const float* __restrict__ W1, bf16* __restrict__ wt) {
  int i = blockIdx.x * blockDim.x + threadIdx.x;   // 512*256
  if (i >= 512 * 256) return;
  int k = i / 256, c = i % 256;
  wt[c * 512 + k] = (bf16)W1[i];
}

// W2 [256,40] f32 -> w2bt[c*256 + k] bf16, c padded to 48 with zeros
__global__ void prep_w2(const float* __restrict__ W2, bf16* __restrict__ wt) {
  int i = blockIdx.x * blockDim.x + threadIdx.x;   // 48*256
  if (i >= 48 * 256) return;
  int c = i / 256, k = i % 256;
  wt[i] = (c < 40) ? (bf16)W2[k * 40 + c] : (bf16)0.0f;
}

__global__ void fill_u32(unsigned* __restrict__ p, long long n, unsigned v) {
  long long i = blockIdx.x * (long long)blockDim.x + threadIdx.x;
  if (i < n) p[i] = v;
}

// ---------------------------------------------------------------------------
// GEMM1: h1[N,256] = xbf[N,512] @ W1  (bf16 WMMA, f32 accum)
// block = 256 threads (8 waves); block computes 16 rows x 256 cols,
// wave w owns column tiles {2w, 2w+1}; K loop = 16 x (16x16x32 wmma) per tile
// ---------------------------------------------------------------------------
__global__ void __launch_bounds__(256)
gemm1_wmma(const bf16* __restrict__ xbf, const bf16* __restrict__ w1bt,
           float* __restrict__ h1, int N) {
  const int lane = threadIdx.x & 31;
  const int wave = threadIdx.x >> 5;
  const int lm = lane & 15, hi = lane >> 4;
  const int mt = blockIdx.x;
  int arow = mt * 16 + lm; if (arow >= N) arow = N - 1;
  const bf16* ap  = xbf + (size_t)arow * 512;
  const int c0 = wave * 2, c1 = wave * 2 + 1;
  const bf16* bp0 = w1bt + (size_t)(c0 * 16 + lm) * 512;
  const bf16* bp1 = w1bt + (size_t)(c1 * 16 + lm) * 512;
  v8f acc0 = {}; v8f acc1 = {};
#pragma unroll
  for (int kb = 0; kb < 512; kb += 32) {
    v16bf a  = load_a_frag(ap,  kb, hi);
    v16bf b0 = load_b_frag(bp0, kb, hi);
    v16bf b1 = load_b_frag(bp1, kb, hi);
    acc0 = __builtin_amdgcn_wmma_f32_16x16x32_bf16(false, a, false, b0,
                                                   (short)0, acc0, false, false);
    acc1 = __builtin_amdgcn_wmma_f32_16x16x32_bf16(false, a, false, b1,
                                                   (short)0, acc1, false, false);
  }
#pragma unroll
  for (int r = 0; r < 8; r++) {              // D row M = r + hi*8
    int orow = mt * 16 + hi * 8 + r;
    if (orow < N) {
      h1[(size_t)orow * 256 + c0 * 16 + lm] = acc0[r];
      h1[(size_t)orow * 256 + c1 * 16 + lm] = acc1[r];
    }
  }
}

// ---------------------------------------------------------------------------
// GEMM2: g2[N,40] = h2elu_bf[N,256] @ W2  (cols padded to 48 -> 3 waves)
// ---------------------------------------------------------------------------
__global__ void __launch_bounds__(128)
gemm2_wmma(const bf16* __restrict__ hbf, const bf16* __restrict__ w2bt,
           float* __restrict__ g2, int N) {
  const int lane = threadIdx.x & 31;
  const int wave = threadIdx.x >> 5;
  if (wave >= 3) return;                     // 3 col tiles of 16 = 48 cols
  const int lm = lane & 15, hi = lane >> 4;
  const int mt = blockIdx.x;
  int arow = mt * 16 + lm; if (arow >= N) arow = N - 1;
  const bf16* ap = hbf + (size_t)arow * 256;
  const bf16* bp = w2bt + (size_t)(wave * 16 + lm) * 256;
  v8f acc = {};
#pragma unroll
  for (int kb = 0; kb < 256; kb += 32) {
    v16bf a = load_a_frag(ap, kb, hi);
    v16bf b = load_b_frag(bp, kb, hi);
    acc = __builtin_amdgcn_wmma_f32_16x16x32_bf16(false, a, false, b,
                                                  (short)0, acc, false, false);
  }
  int col = wave * 16 + lm;
  if (col < 40) {
#pragma unroll
    for (int r = 0; r < 8; r++) {
      int orow = mt * 16 + hi * 8 + r;
      if (orow < N) g2[(size_t)orow * 40 + col] = acc[r];
    }
  }
}

// ---------------------------------------------------------------------------
// Attention scalar dots: a_src[n,h] = sum_c feat[n,h,c]*att_s[h,c] (ditto dst)
// ---------------------------------------------------------------------------
__global__ void att_dots(const float* __restrict__ feat,
                         const float* __restrict__ att_s,
                         const float* __restrict__ att_d,
                         float* __restrict__ as_, float* __restrict__ ad_,
                         int N, int H, int C) {
  int i = blockIdx.x * blockDim.x + threadIdx.x;   // [0, N*H)
  if (i >= N * H) return;
  int n = i / H, h = i % H;
  const float* f = feat + (size_t)n * H * C + (size_t)h * C;
  float s = 0.f, d = 0.f;
  for (int c = 0; c < C; c++) {
    float v = f[c];
    s += v * att_s[h * C + c];
    d += v * att_d[h * C + c];
  }
  as_[i] = s; ad_[i] = d;
}

// ---------------------------------------------------------------------------
// Edge pass 1: segment-max of leaky_relu(a_src[s]+a_dst[d]) per (dst, head)
// ---------------------------------------------------------------------------
template <int H>
__global__ void edge_max_k(const long long* __restrict__ ei, int E, int N,
                           const float* __restrict__ as_,
                           const float* __restrict__ ad_,
                           unsigned* __restrict__ amax) {
  long long t = blockIdx.x * (long long)blockDim.x + threadIdx.x;
  if (t >= (long long)E + N) return;
  int s, d;
  if (t < E) { s = (int)ei[t]; d = (int)ei[E + t]; }
  else       { s = d = (int)(t - E); }               // self loop
#pragma unroll
  for (int h = 0; h < H; h++) {
    float a = as_[s * H + h] + ad_[d * H + h];
    a = a > 0.f ? a : NEG_SLOPE * a;
    atomicMax(&amax[d * H + h], fenc(a));
  }
}

// ---------------------------------------------------------------------------
// Edge pass 2: ex = exp(alpha - max); acc[d,:] += ex*feat[s,:]; denom[d,h]+=ex
// One block per edge; H*C threads (256 for layer1, 40-of-64 for layer2).
// ---------------------------------------------------------------------------
template <int H, int C>
__global__ void edge_msg_k(const long long* __restrict__ ei, int E, int N,
                           const float* __restrict__ as_,
                           const float* __restrict__ ad_,
                           const unsigned* __restrict__ amaxu,
                           const float* __restrict__ feat,
                           float* __restrict__ acc, float* __restrict__ denom) {
  int e = blockIdx.x;
  int s, d;
  if (e < E) { s = (int)ei[e]; d = (int)ei[E + e]; }
  else       { s = d = e - E; }
  int t = threadIdx.x;
  if (t >= H * C) return;
  int h = t / C, c = t % C;
  float a = as_[s * H + h] + ad_[d * H + h];
  a = a > 0.f ? a : NEG_SLOPE * a;
  float ex = __expf(a - fdec(amaxu[d * H + h]));
  atomicAdd(&acc[(size_t)d * (H * C) + t], ex * feat[(size_t)s * (H * C) + t]);
  if (c == 0) atomicAdd(&denom[d * H + h], ex);
}

// ---------------------------------------------------------------------------
// Layer-1 epilogue: normalize, +bias, ELU, cast to bf16 for GEMM2
// ---------------------------------------------------------------------------
__global__ void norm_elu1(const float* __restrict__ acc,
                          const float* __restrict__ denom,
                          const float* __restrict__ bias,
                          bf16* __restrict__ out, int N) {
  long long i = blockIdx.x * (long long)blockDim.x + threadIdx.x;
  if (i >= (long long)N * 256) return;
  int n = (int)(i >> 8);
  int t = (int)(i & 255);
  int h = t >> 6;
  float v = acc[i] / (denom[n * 4 + h] + GAT_EPS) + bias[t];
  v = v > 0.f ? v : (__expf(v) - 1.0f);       // ELU
  out[i] = (bf16)v;
}

// ---------------------------------------------------------------------------
// Layer-2 epilogue: normalize, +bias, log_softmax over 40 classes
// ---------------------------------------------------------------------------
__global__ void final_k(const float* __restrict__ acc2,
                        const float* __restrict__ den2,
                        const float* __restrict__ b2,
                        float* __restrict__ out, int N) {
  int n = blockIdx.x * blockDim.x + threadIdx.x;
  if (n >= N) return;
  float inv = 1.0f / (den2[n] + GAT_EPS);
  float v[40];
  float m = -3.402823e38f;
#pragma unroll
  for (int j = 0; j < 40; j++) {
    v[j] = acc2[(size_t)n * 40 + j] * inv + b2[j];
    m = fmaxf(m, v[j]);
  }
  float s = 0.f;
#pragma unroll
  for (int j = 0; j < 40; j++) s += __expf(v[j] - m);
  float ls = __logf(s);
#pragma unroll
  for (int j = 0; j < 40; j++) out[(size_t)n * 40 + j] = v[j] - m - ls;
}

// ---------------------------------------------------------------------------
// Host-side launch
// ---------------------------------------------------------------------------
extern "C" void kernel_launch(void* const* d_in, const int* in_sizes, int n_in,
                              void* d_out, int out_size, void* d_ws, size_t ws_size,
                              hipStream_t stream) {
  const float*     x    = (const float*)d_in[0];
  const long long* ei   = (const long long*)d_in[1];   // int64 edge_index [2,E]
  const float*     W1   = (const float*)d_in[2];
  const float*     as1  = (const float*)d_in[3];
  const float*     ad1  = (const float*)d_in[4];
  const float*     b1   = (const float*)d_in[5];
  const float*     W2   = (const float*)d_in[6];
  const float*     as2  = (const float*)d_in[7];
  const float*     ad2  = (const float*)d_in[8];
  const float*     b2   = (const float*)d_in[9];
  float* out = (float*)d_out;

  const int N = in_sizes[0] / 512;
  const int E = in_sizes[1] / 2;
  const long long NE = (long long)E + N;

  char* ws = (char*)d_ws;
  // Region A (N*1024 B): xbf (bf16 N*512) -> acc1 (f32 N*256) -> g2 + acc2
  // Region B (N*1024 B): h1 (f32 N*256)   -> h2elu (bf16 N*256)
  const size_t RA = 0;
  const size_t RB = (size_t)N * 1024;
  const size_t SM = 2 * (size_t)N * 1024;
  bf16*  xbf   = (bf16*)(ws + RA);
  float* acc1  = (float*)(ws + RA);
  float* g2    = (float*)(ws + RA);
  float* acc2  = (float*)(ws + RA + (size_t)N * 256);   // byte offset N*256, clear of g2 (N*160 B)
  float* h1    = (float*)(ws + RB);
  bf16*  h2elu = (bf16*)(ws + RB);
  bf16*  w1bt  = (bf16*)(ws + SM);                       // 512*256*2 = 262144 B
  bf16*  w2bt  = (bf16*)(ws + SM + 262144);              // 48*256*2  =  24576 B
  size_t so = SM + 262144 + 24576;
  float*    asrc1 = (float*)(ws + so);    so += (size_t)N * 16;
  float*    adst1 = (float*)(ws + so);    so += (size_t)N * 16;
  unsigned* amax1 = (unsigned*)(ws + so); so += (size_t)N * 16;
  float*    den1  = (float*)(ws + so);    so += (size_t)N * 16;
  float*    asrc2 = (float*)(ws + so);    so += (size_t)N * 4;
  float*    adst2 = (float*)(ws + so);    so += (size_t)N * 4;
  unsigned* amax2 = (unsigned*)(ws + so); so += (size_t)N * 4;
  float*    den2  = (float*)(ws + so);    so += (size_t)N * 4;

  const int T = 256;
  auto blocks = [](long long n, int t) { return (unsigned)((n + t - 1) / t); };

  // ---- prep: bf16 weights (transposed) and bf16 x ----
  prep_w1<<<blocks(512 * 256, T), T, 0, stream>>>(W1, w1bt);
  prep_w2<<<blocks(48 * 256, T), T, 0, stream>>>(W2, w2bt);
  cvt_f32_bf16<<<blocks((long long)N * 512, T), T, 0, stream>>>(x, xbf, (long long)N * 512);

  // ---- layer 1 ----
  gemm1_wmma<<<(N + 15) / 16, 256, 0, stream>>>(xbf, w1bt, h1, N);
  att_dots<<<blocks((long long)N * 4, T), T, 0, stream>>>(h1, as1, ad1, asrc1, adst1, N, 4, 64);
  // region A now free of xbf -> becomes acc1
  fill_u32<<<blocks((long long)N * 256, T), T, 0, stream>>>((unsigned*)acc1, (long long)N * 256, 0u);
  fill_u32<<<blocks((long long)N * 4, T), T, 0, stream>>>((unsigned*)den1, (long long)N * 4, 0u);
  fill_u32<<<blocks((long long)N * 4, T), T, 0, stream>>>(amax1, (long long)N * 4, NEG_INF_KEY);
  edge_max_k<4><<<blocks(NE, T), T, 0, stream>>>(ei, E, N, asrc1, adst1, amax1);
  edge_msg_k<4, 64><<<(unsigned)NE, 256, 0, stream>>>(ei, E, N, asrc1, adst1, amax1, h1, acc1, den1);
  norm_elu1<<<blocks((long long)N * 256, T), T, 0, stream>>>(acc1, den1, b1, h2elu, N);

  // ---- layer 2 (region A now free of acc1 -> g2 + acc2) ----
  fill_u32<<<blocks((long long)N * 40, T), T, 0, stream>>>((unsigned*)acc2, (long long)N * 40, 0u);
  fill_u32<<<blocks((long long)N, T), T, 0, stream>>>((unsigned*)den2, (long long)N, 0u);
  fill_u32<<<blocks((long long)N, T), T, 0, stream>>>(amax2, (long long)N, NEG_INF_KEY);
  gemm2_wmma<<<(N + 15) / 16, 128, 0, stream>>>(h2elu, w2bt, g2, N);
  att_dots<<<blocks((long long)N, T), T, 0, stream>>>(g2, as2, ad2, asrc2, adst2, N, 1, 40);
  edge_max_k<1><<<blocks(NE, T), T, 0, stream>>>(ei, E, N, asrc2, adst2, amax2);
  edge_msg_k<1, 40><<<(unsigned)NE, 64, 0, stream>>>(ei, E, N, asrc2, adst2, amax2, g2, acc2, den2);
  final_k<<<blocks((long long)N, T), T, 0, stream>>>(acc2, den2, b2, out, N);
}
